// AdaptiveGaussKronrod_57887569216040
// MI455X (gfx1250) — matrix operand order repaired and checked by reading, not compiled, not run on hardware
//
#include <hip/hip_runtime.h>
#include <hip/hip_bf16.h>

typedef __attribute__((ext_vector_type(8)))  _Float16 v8h;
typedef __attribute__((ext_vector_type(16))) _Float16 v16h;
typedef __attribute__((ext_vector_type(8)))  float    v8f;

#define D_DIM 4096
#define N_ROWS 1920   /* S(128) * 15 GK nodes */

// Gauss-Kronrod 15-point nodes/weights in the reference's concatenated order.
__constant__ float GKN[15] = {
   0.20778495500789848f,  0.4058451513773972f,  0.5860872354676911f,
   0.7415311855993945f,   0.8648644233597691f,  0.9491079123427585f,
   0.9914553711208126f,
  -0.9914553711208126f,  -0.9491079123427585f, -0.8648644233597691f,
  -0.7415311855993945f,  -0.5860872354676911f, -0.4058451513773972f,
  -0.20778495500789848f,  0.0f };
__constant__ float GKW[15] = {
   0.20443294007529889f,  0.19035057806478542f, 0.1690047266392679f,
   0.14065325971552592f,  0.10479001032225019f, 0.06309209262997856f,
   0.022935322010529224f,
   0.022935322010529224f, 0.06309209262997856f, 0.10479001032225019f,
   0.14065325971552592f,  0.1690047266392679f,  0.19035057806478542f,
   0.20443294007529889f,  0.20948214108472782f };

// ---------------------------------------------------------------- prep: t, scale
__global__ __launch_bounds__(256)
void prep_t_kernel(float* __restrict__ t, float* __restrict__ scale) {
  int n = blockIdx.x * blockDim.x + threadIdx.x;
  if (n >= N_ROWS) return;
  int s = n / 15, k = n % 15;
  const float h = 1.0f / 256.0f;                 // (b-a)/2 for S=128 on [0,1]
  float c = (2.0f * (float)s + 1.0f) / 256.0f;   // segment midpoint
  t[n]     = c + h * GKN[k];
  scale[n] = h * GKW[k];
}

// ---------------------- phi row-major: each thread computes 8 d's, one v8h store
__global__ __launch_bounds__(256)
void phi_kernel(const float* __restrict__ t, const float* __restrict__ freqs,
                _Float16* __restrict__ phi) {
  int id = blockIdx.x * blockDim.x + threadIdx.x;   // N_ROWS * 512 threads
  int n  = id >> 9;
  int d0 = (id & 511) * 8;
  float tv = t[n];
  v8h o;
#pragma unroll
  for (int i = 0; i < 8; ++i)
    o[i] = (_Float16)__builtin_sinf(tv * freqs[d0 + i]);
  *(v8h*)(phi + (size_t)n * D_DIM + d0) = o;
}

// ---------------- phiT row-major (recompute sin in transposed order, coalesced)
__global__ __launch_bounds__(256)
void phiT_kernel(const float* __restrict__ t, const float* __restrict__ freqs,
                 _Float16* __restrict__ phiT) {
  int id = blockIdx.x * blockDim.x + threadIdx.x;   // D_DIM * 240 threads
  int d  = id / 240;
  int n0 = (id % 240) * 8;
  float fv = freqs[d];
  v8h o;
#pragma unroll
  for (int i = 0; i < 8; ++i)
    o[i] = (_Float16)__builtin_sinf(t[n0 + i] * fv);
  *(v8h*)(phiT + (size_t)d * N_ROWS + n0) = o;
}

// ------------- W (f32, row major) -> W^T (f16) via LDS 32x32 tile transpose
__global__ __launch_bounds__(256)
void wht_kernel(const float* __restrict__ W, _Float16* __restrict__ whT) {
  __shared__ _Float16 tile[32][33];                 // +1 pad: no bank conflicts
  const int tx = threadIdx.x;                       // 0..31
  const int ty = threadIdx.y;                       // 0..7
  const int k0 = blockIdx.y * 32;                   // W row block (contraction)
  const int j0 = blockIdx.x * 32;                   // W col block
#pragma unroll
  for (int i = 0; i < 4; ++i) {                     // coalesced f32 reads
    int r = ty + 8 * i;
    tile[r][tx] = (_Float16)W[(size_t)(k0 + r) * D_DIM + (j0 + tx)];
  }
  __syncthreads();
#pragma unroll
  for (int i = 0; i < 4; ++i) {                     // coalesced f16 writes
    int r = ty + 8 * i;
    whT[(size_t)(j0 + r) * D_DIM + (k0 + tx)] = tile[tx][r];
  }
}

// ---------------------------------------------------------------- helpers
__device__ __forceinline__ v16h make_a_frag(const _Float16* p) {
  // per-lane A fragment: 16 contiguous halves (K region of this lane half)
  v8h lo = *(const v8h*)(p);
  v8h hi = *(const v8h*)(p + 8);
  return __builtin_shufflevector(lo, hi, 0,1,2,3,4,5,6,7,8,9,10,11,12,13,14,15);
}

// ============ GEMM1: u = phi @ W + b ; gT[j][n] = scale*cos(t*af)*(1-tanh(u)^2)
// Block = 8 waves in a 2(M) x 4(N) grid -> 64 x 256 block tile.
// Each wave owns a 32(M=n) x 64(N=j) strip; A/B f16, f32 accumulate.
__global__ __launch_bounds__(256)
void gemm1_kernel(const _Float16* __restrict__ phi, const _Float16* __restrict__ whT,
                  const float* __restrict__ bvec,  const float* __restrict__ afreqs,
                  const float* __restrict__ t,     const float* __restrict__ scale,
                  _Float16* __restrict__ gT) {
  const int wave = threadIdx.x >> 5;
  const int wm = wave >> 2;        // 0..1 : M position in block
  const int wn = wave & 3;         // 0..3 : N position in block
  const int lane = threadIdx.x & 31;
  const int lo = lane & 15;        // A: M row  | B/C: N col
  const int hi = lane >> 4;        // K-half selector / C M-half
  const int tile_n  = blockIdx.y * 64  + wm * 32;      // rows (quadrature dim)
  const int tile_j0 = blockIdx.x * 256 + wn * 64;      // cols (feature dim)

  const _Float16* arow0 = phi + (size_t)(tile_n +  0 + lo) * D_DIM + hi * 16;
  const _Float16* arow1 = phi + (size_t)(tile_n + 16 + lo) * D_DIM + hi * 16;
  const _Float16* brow0 = whT + (size_t)(tile_j0 +  0 + lo) * D_DIM + hi * 16;
  const _Float16* brow1 = whT + (size_t)(tile_j0 + 16 + lo) * D_DIM + hi * 16;
  const _Float16* brow2 = whT + (size_t)(tile_j0 + 32 + lo) * D_DIM + hi * 16;
  const _Float16* brow3 = whT + (size_t)(tile_j0 + 48 + lo) * D_DIM + hi * 16;

  v8f c00 = {}, c01 = {}, c02 = {}, c03 = {};
  v8f c10 = {}, c11 = {}, c12 = {}, c13 = {};
  for (int k0 = 0; k0 < D_DIM; k0 += 32) {
    __builtin_prefetch(arow0 + k0 + 64, 0, 3);   // global_prefetch_b8
    __builtin_prefetch(brow0 + k0 + 64, 0, 3);
    v16h a0 = make_a_frag(arow0 + k0);
    v16h a1 = make_a_frag(arow1 + k0);
    v16h b0 = *(const v16h*)(brow0 + k0);
    v16h b1 = *(const v16h*)(brow1 + k0);
    v16h b2 = *(const v16h*)(brow2 + k0);
    v16h b3 = *(const v16h*)(brow3 + k0);
    c00 = __builtin_amdgcn_wmma_f32_16x16x32_f16(false, a0, false, b0, (short)0, c00, false, false);
    c01 = __builtin_amdgcn_wmma_f32_16x16x32_f16(false, a0, false, b1, (short)0, c01, false, false);
    c02 = __builtin_amdgcn_wmma_f32_16x16x32_f16(false, a0, false, b2, (short)0, c02, false, false);
    c03 = __builtin_amdgcn_wmma_f32_16x16x32_f16(false, a0, false, b3, (short)0, c03, false, false);
    c10 = __builtin_amdgcn_wmma_f32_16x16x32_f16(false, a1, false, b0, (short)0, c10, false, false);
    c11 = __builtin_amdgcn_wmma_f32_16x16x32_f16(false, a1, false, b1, (short)0, c11, false, false);
    c12 = __builtin_amdgcn_wmma_f32_16x16x32_f16(false, a1, false, b2, (short)0, c12, false, false);
    c13 = __builtin_amdgcn_wmma_f32_16x16x32_f16(false, a1, false, b3, (short)0, c13, false, false);
  }

  // Epilogue. C layout: lane holds column (lo), rows M = hi*8 + r, r=0..7.
  v8f cc[2][4] = { { c00, c01, c02, c03 }, { c10, c11, c12, c13 } };
#pragma unroll
  for (int mt = 0; mt < 2; ++mt) {
    const int nbase = tile_n + mt * 16 + hi * 8;
    float ts[8], ss[8];
#pragma unroll
    for (int r = 0; r < 8; ++r) { ts[r] = t[nbase + r]; ss[r] = scale[nbase + r]; }
#pragma unroll
    for (int jt = 0; jt < 4; ++jt) {
      int j = tile_j0 + jt * 16 + lo;
      float bj = bvec[j];
      float af = afreqs[j];
      v8h outp;
#pragma unroll
      for (int r = 0; r < 8; ++r) {
        float u  = cc[mt][jt][r] + bj;
        float y  = tanhf(u);
        float gv = ss[r] * __builtin_cosf(ts[r] * af) * (1.0f - y * y);
        outp[r] = (_Float16)gv;
      }
      // transposed store: per lane, 8 contiguous f16 along n -> one b128 store
      *(v8h*)(gT + (size_t)j * N_ROWS + nbase) = outp;
    }
  }
}

// ============ GEMM2: out[i][j] = sum_n phiT[i][n] * gT[j][n]   (fp32 out)
__global__ __launch_bounds__(256)
void gemm2_kernel(const _Float16* __restrict__ phiT, const _Float16* __restrict__ gT,
                  float* __restrict__ out) {
  const int wave = threadIdx.x >> 5;
  const int wm = wave >> 2;
  const int wn = wave & 3;
  const int lane = threadIdx.x & 31;
  const int lo = lane & 15;
  const int hi = lane >> 4;
  const int tile_i  = blockIdx.y * 64  + wm * 32;
  const int tile_j0 = blockIdx.x * 256 + wn * 64;

  const _Float16* arow0 = phiT + (size_t)(tile_i +  0 + lo) * N_ROWS + hi * 16;
  const _Float16* arow1 = phiT + (size_t)(tile_i + 16 + lo) * N_ROWS + hi * 16;
  const _Float16* brow0 = gT + (size_t)(tile_j0 +  0 + lo) * N_ROWS + hi * 16;
  const _Float16* brow1 = gT + (size_t)(tile_j0 + 16 + lo) * N_ROWS + hi * 16;
  const _Float16* brow2 = gT + (size_t)(tile_j0 + 32 + lo) * N_ROWS + hi * 16;
  const _Float16* brow3 = gT + (size_t)(tile_j0 + 48 + lo) * N_ROWS + hi * 16;

  v8f c00 = {}, c01 = {}, c02 = {}, c03 = {};
  v8f c10 = {}, c11 = {}, c12 = {}, c13 = {};
  for (int k0 = 0; k0 < N_ROWS; k0 += 32) {
    __builtin_prefetch(arow0 + k0 + 64, 0, 3);
    __builtin_prefetch(brow0 + k0 + 64, 0, 3);
    v16h a0 = make_a_frag(arow0 + k0);
    v16h a1 = make_a_frag(arow1 + k0);
    v16h b0 = *(const v16h*)(brow0 + k0);
    v16h b1 = *(const v16h*)(brow1 + k0);
    v16h b2 = *(const v16h*)(brow2 + k0);
    v16h b3 = *(const v16h*)(brow3 + k0);
    c00 = __builtin_amdgcn_wmma_f32_16x16x32_f16(false, a0, false, b0, (short)0, c00, false, false);
    c01 = __builtin_amdgcn_wmma_f32_16x16x32_f16(false, a0, false, b1, (short)0, c01, false, false);
    c02 = __builtin_amdgcn_wmma_f32_16x16x32_f16(false, a0, false, b2, (short)0, c02, false, false);
    c03 = __builtin_amdgcn_wmma_f32_16x16x32_f16(false, a0, false, b3, (short)0, c03, false, false);
    c10 = __builtin_amdgcn_wmma_f32_16x16x32_f16(false, a1, false, b0, (short)0, c10, false, false);
    c11 = __builtin_amdgcn_wmma_f32_16x16x32_f16(false, a1, false, b1, (short)0, c11, false, false);
    c12 = __builtin_amdgcn_wmma_f32_16x16x32_f16(false, a1, false, b2, (short)0, c12, false, false);
    c13 = __builtin_amdgcn_wmma_f32_16x16x32_f16(false, a1, false, b3, (short)0, c13, false, false);
  }

  v8f cc[2][4] = { { c00, c01, c02, c03 }, { c10, c11, c12, c13 } };
#pragma unroll
  for (int mt = 0; mt < 2; ++mt) {
#pragma unroll
    for (int jt = 0; jt < 4; ++jt) {
      int j = tile_j0 + jt * 16 + lo;
#pragma unroll
      for (int r = 0; r < 8; ++r) {
        // lanes 0..15 of a half-wave write 16 consecutive dwords of one row
        out[(size_t)(tile_i + mt * 16 + hi * 8 + r) * D_DIM + j] = cc[mt][jt][r];
      }
    }
  }
}

// ---------------------------------------------------------------- launcher
extern "C" void kernel_launch(void* const* d_in, const int* in_sizes, int n_in,
                              void* d_out, int out_size, void* d_ws, size_t ws_size,
                              hipStream_t stream) {
  (void)in_sizes; (void)n_in; (void)out_size; (void)ws_size;
  const float* W      = (const float*)d_in[0];
  const float* bvec   = (const float*)d_in[1];
  const float* freqs  = (const float*)d_in[2];
  const float* afreqs = (const float*)d_in[3];
  float* out = (float*)d_out;

  char* ws = (char*)d_ws;
  const size_t PHI_BYTES = (size_t)N_ROWS * D_DIM * sizeof(_Float16);  // 15.7 MB
  const size_t WHT_BYTES = (size_t)D_DIM * D_DIM * sizeof(_Float16);   // 33.5 MB
  float*    t     = (float*)(ws + 0);
  float*    scale = (float*)(ws + 8192);
  _Float16* phi   = (_Float16*)(ws + 16384);
  _Float16* phiT  = (_Float16*)(ws + 16384 + PHI_BYTES);
  _Float16* whT   = (_Float16*)(ws + 16384 + 2 * PHI_BYTES);
  _Float16* gT    = (_Float16*)(ws + 16384 + 2 * PHI_BYTES + WHT_BYTES);

  prep_t_kernel<<<(N_ROWS + 255) / 256, 256, 0, stream>>>(t, scale);
  phi_kernel <<<(N_ROWS * 512) / 256, 256, 0, stream>>>(t, freqs, phi);
  phiT_kernel<<<(D_DIM * 240) / 256, 256, 0, stream>>>(t, freqs, phiT);
  wht_kernel<<<dim3(D_DIM / 32, D_DIM / 32), dim3(32, 8), 0, stream>>>(W, whT);
  // GEMM1: 64x256 block tiles -> grid (4096/256, 1920/64) = (16, 30)
  gemm1_kernel<<<dim3(D_DIM / 256, N_ROWS / 64), 256, 0, stream>>>(
      phi, whT, bvec, afreqs, t, scale, gT);
  // GEMM2: 64x256 block tiles -> grid (16, 4096/64) = (16, 64)
  gemm2_kernel<<<dim3(D_DIM / 256, D_DIM / 64), 256, 0, stream>>>(phiT, gT, out);
}
